// SlidingWindowAttention_90555090469073
// MI455X (gfx1250) — matrix operand order repaired and checked by reading, not compiled, not run on hardware
//
#include <hip/hip_runtime.h>
#include <hip/hip_bf16.h>

typedef _Float16 half_t;
typedef __attribute__((ext_vector_type(16))) _Float16 v16h;
typedef __attribute__((ext_vector_type(8)))  _Float16 v8h;
typedef __attribute__((ext_vector_type(8)))  float    v8f;

#define SEQ_N   2048
#define DMODEL  1024
#define NHEADS  16
#define DHEAD   64
#define WINDOW  128
#define BATCH   2
#define MTOT    (BATCH * SEQ_N)   // 4096 rows for the big GEMMs

static __device__ __forceinline__ v16h cat8(v8h lo, v8h hi) {
    return __builtin_shufflevector(lo, hi, 0,1,2,3,4,5,6,7,8,9,10,11,12,13,14,15);
}

// ---------------------------------------------------------------------------
// Stage 0a: f32 -> f16 elementwise convert (x)
// ---------------------------------------------------------------------------
__global__ void swa_convert_f16(const float* __restrict__ src,
                                half_t* __restrict__ dst, int n) {
    int i = blockIdx.x * blockDim.x + threadIdx.x;
    if (i < n) dst[i] = (half_t)src[i];
}

// ---------------------------------------------------------------------------
// Stage 0b: W[n][k] f32 -> Wt[k][n] f16 (1024x1024), LDS-tiled transpose
// ---------------------------------------------------------------------------
__global__ void swa_transpose_f16(const float* __restrict__ src,
                                  half_t* __restrict__ dst) {
    __shared__ float tile[32][33];
    int x = blockIdx.x * 32 + threadIdx.x;   // k (input col)
    int y0 = blockIdx.y * 32;                // n (input row base)
    #pragma unroll
    for (int j = threadIdx.y; j < 32; j += 8)
        tile[j][threadIdx.x] = src[(size_t)(y0 + j) * DMODEL + x];
    __syncthreads();
    int xo = blockIdx.y * 32 + threadIdx.x;  // n (output col)
    int yo0 = blockIdx.x * 32;               // k (output row base)
    #pragma unroll
    for (int j = threadIdx.y; j < 32; j += 8)
        dst[(size_t)(yo0 + j) * DMODEL + xo] = (half_t)tile[threadIdx.x][j];
}

// ---------------------------------------------------------------------------
// WMMA GEMM: C(4096x1024) = A(4096x1024,f16) @ Bt(1024x1024,f16) + bias
// Bt is pre-transposed so Bt[k][n] = W[n][k]  (i.e. computes A @ W^T + b).
// One wave computes a 16x64 tile.  MODE: 0 = f16 out [b][h][n][d] (Q/V),
// 1 = f16 out [b][h][d][n] (K transposed), 2 = f32 out row-major (final).
// ---------------------------------------------------------------------------
template <int MODE>
__global__ void swa_gemm_wmma(const half_t* __restrict__ A,
                              const half_t* __restrict__ Bt,
                              const float* __restrict__ bias,
                              void* __restrict__ out) {
    const int lane = threadIdx.x & 31;
    // wave-uniform tile id, forced into SGPRs
    const int w  = __builtin_amdgcn_readfirstlane(blockIdx.x * 8 + (threadIdx.x >> 5));
    const int mt = w >> 4;                            // 0..255
    const int nt = w & 15;                            // 0..15
    const int m0 = mt * 16;
    const int n0 = nt * 64;
    const int hs   = lane >> 4;                       // lane half (0/1)
    const int l15  = lane & 15;

    v8f acc[4] = {{}, {}, {}, {}};

    const size_t arow = (size_t)(m0 + l15) * DMODEL;
    for (int k0 = 0; k0 < DMODEL; k0 += 32) {
        const int kb = k0 + hs * 8;
        v8h alo = *(const v8h*)(A + arow + kb);
        v8h ahi = *(const v8h*)(A + arow + kb + 16);
        v16h a = cat8(alo, ahi);
        v16h b[4];
        #pragma unroll
        for (int nc = 0; nc < 4; ++nc)
            #pragma unroll
            for (int s = 0; s < 16; ++s)
                b[nc][s] = Bt[(size_t)(kb + (s & 7) + ((s >> 3) << 4)) * DMODEL + n0 + nc * 16 + l15];
        #pragma unroll
        for (int nc = 0; nc < 4; ++nc)
            acc[nc] = __builtin_amdgcn_wmma_f32_16x16x32_f16(
                false, a, false, b[nc], (short)0, acc[nc], false, false);
    }

    #pragma unroll
    for (int nc = 0; nc < 4; ++nc) {
        const int nfeat = n0 + nc * 16 + l15;
        const float bv = bias[nfeat];
        #pragma unroll
        for (int v = 0; v < 8; ++v) {
            const int m = m0 + v + 8 * hs;
            const float val = acc[nc][v] + bv;
            if (MODE == 2) {
                ((float*)out)[(size_t)m * DMODEL + nfeat] = val;
            } else {
                const int b_   = m >> 11;       // / SEQ_N
                const int nseq = m & (SEQ_N - 1);
                const int h    = nfeat >> 6;    // / DHEAD
                const int d    = nfeat & (DHEAD - 1);
                size_t idx;
                if (MODE == 0)   // [b][h][n][d]
                    idx = ((size_t)(b_ * NHEADS + h) * SEQ_N + nseq) * DHEAD + d;
                else             // [b][h][d][n]
                    idx = ((size_t)(b_ * NHEADS + h) * DHEAD + d) * SEQ_N + nseq;
                ((half_t*)out)[idx] = (half_t)val;
            }
        }
    }
}

// ---------------------------------------------------------------------------
// Flash-style sliding-window attention.
// One wave per (b,h,16-query tile).  Per 32-key pair: 4 WMMAs for S=Q K^T,
// online softmax, P re-layout via per-wave LDS, 4 WMMAs for P V.
// Control flow is wave-uniform scalar (readfirstlane) => EXEC all-ones for
// every WMMA.  Band mask: |i-j|<=W  <=>  (unsigned)(j-i+W) <= 2W, one
// compare per element with a hoisted per-lane base.  Masked scores are
// -3e38; since every (row, key-pair) contains at least one in-band key,
// mnew is always a real score and __expf(-3e38 - mnew) underflows to 0,
// so no explicit guard is needed on p or alpha.
// Qh: [b][h][n][d] f16;  Ktr: [b][h][d][n] f16;  Vh: [b][h][n][d] f16;
// Oh: [b][n][h][d] f16 (A-layout-friendly for the output projection).
// ---------------------------------------------------------------------------
__global__ void swa_attention_wmma(const half_t* __restrict__ Qh,
                                   const half_t* __restrict__ Ktr,
                                   const half_t* __restrict__ Vh,
                                   half_t* __restrict__ Oh) {
    __shared__ _Float16 Pshared[8][16 * 32];   // per-wave P tile (16x32)

    const int lane = threadIdx.x & 31;
    const int wave = threadIdx.x >> 5;
    // wave-uniform work id, forced scalar
    const int w  = __builtin_amdgcn_readfirstlane(blockIdx.x * 8 + wave);
    const int bh = w >> 7;                     // 0..31  (b*H + h)
    const int qt = w & 127;                    // query tile
    const int q0 = qt * 16;
    const int b  = bh >> 4;
    const int h  = bh & 15;
    const int hs   = lane >> 4;
    const int l15  = lane & 15;

    const half_t* Qb = Qh  + (size_t)bh * SEQ_N * DHEAD;
    const half_t* Kb = Ktr + (size_t)bh * DHEAD * SEQ_N;
    const half_t* Vb = Vh  + (size_t)bh * SEQ_N * DHEAD;
    _Float16* P = &Pshared[wave][0];

    // Load Q tile into A-layout registers (two K=32 chunks cover d=64)
    v16h aq[2];
    #pragma unroll
    for (int kc = 0; kc < 2; ++kc) {
        const half_t* qp = Qb + (size_t)(q0 + l15) * DHEAD + kc * 32 + 8 * hs;
        aq[kc] = cat8(*(const v8h*)qp, *(const v8h*)(qp + 16));
    }

    v8f acc[4] = {{}, {}, {}, {}};
    float mrow[8], lrow[8];
    #pragma unroll
    for (int v = 0; v < 8; ++v) { mrow[v] = -3.0e38f; lrow[v] = 0.0f; }

    const int klo  = (q0 - WINDOW) > 0 ? (q0 - WINDOW) : 0;
    const int kend = (q0 + 16 + WINDOW) < SEQ_N ? (q0 + 16 + WINDOW) : SEQ_N;

    // per-lane mask base:  (j - i + WINDOW) for v=0, kt0=0
    const int mbase = l15 + WINDOW - q0 - 8 * hs;

    for (int kv0 = klo; kv0 < kend; kv0 += 32) {   // scalar loop
        // ---- scores for the two 16-key subtiles (branchless) ----
        float sm[2][8];
        #pragma unroll
        for (int st = 0; st < 2; ++st) {
            const int kt0    = kv0 + st * 16;              // actual key base
            const bool active = (kt0 < kend);              // scalar
            const int kt0c   = active ? kt0 : (kend - 16); // safe address base
            v16h bk[2];
            #pragma unroll
            for (int kc = 0; kc < 2; ++kc)
                #pragma unroll
                for (int s = 0; s < 16; ++s)
                    bk[kc][s] = Kb[(size_t)(kc * 32 + 8 * hs + (s & 7) + ((s >> 3) << 4)) * SEQ_N + kt0c + l15];
            v8f sraw = {};
            #pragma unroll
            for (int kc = 0; kc < 2; ++kc)
                sraw = __builtin_amdgcn_wmma_f32_16x16x32_f16(
                    false, aq[kc], false, bk[kc], (short)0, sraw, false, false);
            // dbase = j - i + WINDOW for v=0; inactive -> huge => invalid
            const int dbase = active ? (mbase + kt0) : (int)0x40000000;
            #pragma unroll
            for (int v = 0; v < 8; ++v) {
                const bool valid = (unsigned)(dbase - v) <= (unsigned)(2 * WINDOW);
                sm[st][v] = valid ? sraw[v] * 0.125f : -3.0e38f;
            }
        }

        // ---- online softmax (row spread across 16 lanes of each half) ----
        float alpha[8];
        #pragma unroll
        for (int v = 0; v < 8; ++v) {
            float t = fmaxf(sm[0][v], sm[1][v]);
            #pragma unroll
            for (int off = 1; off < 16; off <<= 1)
                t = fmaxf(t, __shfl_xor(t, off, 32));
            const float mnew = fmaxf(mrow[v], t);   // always a finite score
            const float p0 = __expf(sm[0][v] - mnew);  // underflows to 0 if masked
            const float p1 = __expf(sm[1][v] - mnew);
            float rs = p0 + p1;
            #pragma unroll
            for (int off = 1; off < 16; off <<= 1)
                rs += __shfl_xor(rs, off, 32);
            const float a = __expf(mrow[v] - mnew);    // 0 on first iteration
            lrow[v] = lrow[v] * a + rs;
            mrow[v] = mnew;
            alpha[v] = a;
            const int r = v + 8 * hs;
            P[r * 32 + l15]      = (_Float16)p0;
            P[r * 32 + 16 + l15] = (_Float16)p1;
        }
        asm volatile("s_wait_dscnt 0" ::: "memory");

        #pragma unroll
        for (int nc = 0; nc < 4; ++nc)
            #pragma unroll
            for (int v = 0; v < 8; ++v)
                acc[nc][v] *= alpha[v];

        // ---- P (A-layout from LDS) @ V ----
        const half_t* pp = P + l15 * 32 + 8 * hs;
        v16h ap = cat8(*(const v8h*)pp, *(const v8h*)(pp + 16));
        v16h bv[4];
        #pragma unroll
        for (int s = 0; s < 16; ++s) {
            int kr = kv0 + 8 * hs + (s & 7) + ((s >> 3) << 4);
            if (kr > SEQ_N - 1) kr = SEQ_N - 1;   // clamped rows pair with P==0
            #pragma unroll
            for (int nc = 0; nc < 4; ++nc)
                bv[nc][s] = Vb[(size_t)kr * DHEAD + nc * 16 + l15];
        }
        #pragma unroll
        for (int nc = 0; nc < 4; ++nc)
            acc[nc] = __builtin_amdgcn_wmma_f32_16x16x32_f16(
                false, ap, false, bv[nc], (short)0, acc[nc], false, false);
        asm volatile("" ::: "memory");  // keep next P-stores after these loads
    }

    // ---- normalize and store O as [b][n][h][d] ----
    #pragma unroll
    for (int nc = 0; nc < 4; ++nc) {
        const int d = nc * 16 + l15;
        #pragma unroll
        for (int v = 0; v < 8; ++v) {
            const int q = q0 + v + 8 * hs;
            const float o = acc[nc][v] / lrow[v];
            Oh[(((size_t)b * SEQ_N + q) * NHEADS + h) * DHEAD + d] = (half_t)o;
        }
    }
}

// ---------------------------------------------------------------------------
extern "C" void kernel_launch(void* const* d_in, const int* in_sizes, int n_in,
                              void* d_out, int out_size, void* d_ws, size_t ws_size,
                              hipStream_t stream) {
    const float* x  = (const float*)d_in[0];
    const float* Wq = (const float*)d_in[1];
    const float* bq = (const float*)d_in[2];
    const float* Wk = (const float*)d_in[3];
    const float* bk = (const float*)d_in[4];
    const float* Wv = (const float*)d_in[5];
    const float* bv = (const float*)d_in[6];
    const float* Wo = (const float*)d_in[7];
    const float* bo = (const float*)d_in[8];

    char* ws = (char*)d_ws;
    half_t* xh  = (half_t*)(ws);                          // 8 MB
    half_t* Wqt = (half_t*)(ws + ((size_t)8  << 20));     // 2 MB each
    half_t* Wkt = (half_t*)(ws + ((size_t)10 << 20));
    half_t* Wvt = (half_t*)(ws + ((size_t)12 << 20));
    half_t* Wot = (half_t*)(ws + ((size_t)14 << 20));
    half_t* Qh  = (half_t*)(ws + ((size_t)16 << 20));     // 8 MB
    half_t* Ktr = (half_t*)(ws + ((size_t)24 << 20));     // 8 MB
    half_t* Vh  = (half_t*)(ws + ((size_t)32 << 20));     // 8 MB
    half_t* Oh  = (half_t*)(ws + ((size_t)40 << 20));     // 8 MB

    const int nElem = MTOT * DMODEL;  // 4,194,304
    swa_convert_f16<<<(nElem + 255) / 256, 256, 0, stream>>>(x, xh, nElem);

    dim3 tb(32, 8), tg(DMODEL / 32, DMODEL / 32);
    swa_transpose_f16<<<tg, tb, 0, stream>>>(Wq, Wqt);
    swa_transpose_f16<<<tg, tb, 0, stream>>>(Wk, Wkt);
    swa_transpose_f16<<<tg, tb, 0, stream>>>(Wv, Wvt);
    swa_transpose_f16<<<tg, tb, 0, stream>>>(Wo, Wot);

    // 4096 waves of 16x64 tiles -> 512 blocks x 8 waves
    swa_gemm_wmma<0><<<512, 256, 0, stream>>>(xh, Wqt, bq, (void*)Qh);
    swa_gemm_wmma<1><<<512, 256, 0, stream>>>(xh, Wkt, bk, (void*)Ktr);
    swa_gemm_wmma<0><<<512, 256, 0, stream>>>(xh, Wvt, bv, (void*)Vh);

    // 4096 (b,h,qtile) waves -> 512 blocks
    swa_attention_wmma<<<512, 256, 0, stream>>>(Qh, Ktr, Vh, Oh);

    swa_gemm_wmma<2><<<512, 256, 0, stream>>>(Oh, Wot, bo, d_out);
}
